// RnnModel_78254304133187
// MI455X (gfx1250) — matrix operand order repaired and checked by reading, not compiled, not run on hardware
//
#include <hip/hip_runtime.h>

#define Hh   128
#define Ii   4
#define Bb   256
#define Tt   512
#define BT   16     // batch tile (= WMMA M)
#define KAUG 160    // augmented K: 128 h + 4 x + 1 const + pad (5 chunks of 32)
#define HPAD 168    // padded augmented row (f16 elems): 336B stride rotates banks

typedef __attribute__((ext_vector_type(16))) _Float16 v16h;
typedef __attribute__((ext_vector_type(8)))  _Float16 v8h;
typedef __attribute__((ext_vector_type(4)))  _Float16 v4h;
typedef __attribute__((ext_vector_type(8)))  float    v8f;

__device__ __forceinline__ float fast_tanh(float x) {
#if __has_builtin(__builtin_amdgcn_tanhf)
    return __builtin_amdgcn_tanhf(x);          // v_tanh_f32 (gfx1250 TRANS op)
#else
    const float e = __expf(2.0f * x);          // branchless fallback
    return 1.0f - 2.0f * __builtin_amdgcn_rcpf(e + 1.0f);
#endif
}

__device__ __forceinline__ float fast_sigmoid(float x) {
    return fmaf(0.5f, fast_tanh(0.5f * x), 0.5f);
}

__launch_bounds__(256, 1)
__global__ void bilstm_kernel(const float* __restrict__ x,
                              const float* __restrict__ Wih_f, const float* __restrict__ Whh_f,
                              const float* __restrict__ bih_f, const float* __restrict__ bhh_f,
                              const float* __restrict__ Wih_b, const float* __restrict__ Whh_b,
                              const float* __restrict__ bih_b, const float* __restrict__ bhh_b,
                              float* __restrict__ out)
{
    const int tid   = threadIdx.x;
    const int wv    = tid >> 5;        // wave id 0..7 -> owns h cols wv*16..wv*16+15
    const int lane  = tid & 31;
    const int col   = lane & 15;       // N index inside 16x16 tile
    const int khalf = lane >> 4;       // lane-half for A/B K split
    const int mrow  = khalf << 3;      // row offset of this lane's half in C layout
    const int b0    = blockIdx.x * BT;
    const int dir   = blockIdx.y;

    const float* Wih = dir ? Wih_b : Wih_f;
    const float* Whh = dir ? Whh_b : Whh_f;
    const float* bih = dir ? bih_b : bih_f;
    const float* bhh = dir ? bhh_b : bhh_f;

    // augmented state buffers: row = batch row, cols 0..127 = h_t (f16),
    // cols 128..131 = x_t (f16), col 132 = 1.0 (bias lane), 133..159 = 0
    __shared__ __align__(16) _Float16 hls[2][BT * HPAD];

    // ---- preload augmented B = [Whh^T ; Wih^T ; bias ; 0] into VGPRs ----
    // B 32x16 f16 layout: lanes 0-15 col n=lane hold K=kc*32+0..15;
    //                     lanes 16-31 hold K=kc*32+16..31 (consecutive).
    v16h Bf[4][5];   // [gate tile j][k-chunk kc]
    #pragma unroll
    for (int j = 0; j < 4; ++j) {
        const int g = j * Hh + wv * 16 + col;
        const float* wr = Whh + (size_t)g * Hh;      // Whh[g, :]
        #pragma unroll
        for (int kc = 0; kc < 4; ++kc) {
            const int kb = kc * 32 + khalf * 16;
            v16h b;
            #pragma unroll
            for (int e = 0; e < 16; ++e) b[e] = (_Float16)wr[kb + e];
            Bf[j][kc] = b;
        }
        // kc = 4: k rows 128..131 = Wih[g, 0..3], row 132 = fused bias, rest 0
        {
            v16h b = (v16h)(_Float16)0.0f;
            if (khalf == 0) {
                #pragma unroll
                for (int e = 0; e < Ii; ++e) b[e] = (_Float16)Wih[g * Ii + e];
                b[Ii] = (_Float16)(bih[g] + bhh[g]);
            }
            Bf[j][4] = b;
        }
    }

    // zero both buffers (h0 = 0 and the permanent zero-pad columns)
    {
        unsigned int* z = (unsigned int*)&hls[0][0];
        for (int idx = tid; idx < (2 * BT * HPAD) / 2; idx += 256) z[idx] = 0u;
    }
    __syncthreads();
    if (tid < 2 * BT) {   // constant-1.0 bias column in BOTH buffers
        hls[tid >> 4][(tid & 15) * HPAD + Hh + Ii] = (_Float16)1.0f;
    }
    if (tid < BT) {       // x_0 -> augmented cols of buffer 0
        const int tm = dir ? (Tt - 1) : 0;
        const float4 xv = *(const float4*)(x + ((size_t)(b0 + tid) * Tt + tm) * Ii);
        v4h xh; xh[0] = (_Float16)xv.x; xh[1] = (_Float16)xv.y;
                xh[2] = (_Float16)xv.z; xh[3] = (_Float16)xv.w;
        *(v4h*)&hls[0][tid * HPAD + Hh] = xh;
    }

    v8f cst = (v8f)0.0f;               // cell state tile (16x16 f32, C layout)
    __syncthreads();

    for (int t = 0; t < Tt; ++t) {
        const int p = t & 1;

        // prefetch next step's x tile into the other buffer's augmented cols
        if (tid < BT && t + 1 < Tt) {
            const int tm = dir ? (Tt - 2 - t) : (t + 1);
            const float4 xv = *(const float4*)(x + ((size_t)(b0 + tid) * Tt + tm) * Ii);
            v4h xh; xh[0] = (_Float16)xv.x; xh[1] = (_Float16)xv.y;
                    xh[2] = (_Float16)xv.z; xh[3] = (_Float16)xv.w;
            *(v4h*)&hls[1 - p][tid * HPAD + Hh] = xh;
        }

        // gates = [h_t | x_t | 1] @ [Whh^T ; Wih^T ; bias] via WMMA, C0 = inline 0.
        // A 16x32 layout: lanes 0-15 row m=lane, elems K0..15;
        //                 lanes 16-31 row m=lane-16, elems K16..31 (consecutive).
        v8f acc[4];
        const _Float16* hrow = &hls[p][col * HPAD];
        #pragma unroll
        for (int kc = 0; kc < 5; ++kc) {
            const int kb = kc * 32 + khalf * 16;
            const v8h lo = *(const v8h*)(hrow + kb);
            const v8h hi = *(const v8h*)(hrow + kb + 8);
            const v16h a = __builtin_shufflevector(lo, hi,
                0,1,2,3,4,5,6,7,8,9,10,11,12,13,14,15);
            #pragma unroll
            for (int j = 0; j < 4; ++j) {
                const v8f cin = (kc == 0) ? (v8f)0.0f : acc[j];  // inline-0 C
                acc[j] = __builtin_amdgcn_wmma_f32_16x16x32_f16(
                    false, a, false, Bf[j][kc], (short)0, cin, false, false);
            }
        }

        // LSTM cell update (branchless, v_tanh_f32) + publish h_{t+1} as f16
        _Float16* hn = &hls[1 - p][0];
        #pragma unroll
        for (int v = 0; v < 8; ++v) {
            const float ig = fast_sigmoid(acc[0][v]);
            const float fg = fast_sigmoid(acc[1][v]);
            const float gg = fast_tanh(acc[2][v]);
            const float og = fast_sigmoid(acc[3][v]);
            const float c  = fmaf(fg, cst[v], ig * gg);
            cst[v] = c;
            const float hv = og * fast_tanh(c);
            hn[(mrow + v) * HPAD + wv * 16 + col] = (_Float16)hv;
            if (t == Tt - 1) {   // final hidden state -> output (f32)
                out[(size_t)(b0 + mrow + v) * (2 * Hh) + dir * Hh + wv * 16 + col] = hv;
            }
        }
        __syncthreads();
    }
}

extern "C" void kernel_launch(void* const* d_in, const int* in_sizes, int n_in,
                              void* d_out, int out_size, void* d_ws, size_t ws_size,
                              hipStream_t stream) {
    const float* x     = (const float*)d_in[0];
    const float* Wih_f = (const float*)d_in[1];
    const float* Whh_f = (const float*)d_in[2];
    const float* bih_f = (const float*)d_in[3];
    const float* bhh_f = (const float*)d_in[4];
    const float* Wih_b = (const float*)d_in[5];
    const float* Whh_b = (const float*)d_in[6];
    const float* bih_b = (const float*)d_in[7];
    const float* bhh_b = (const float*)d_in[8];

    dim3 grid(Bb / BT, 2);   // 16 batch tiles x 2 directions = 32 independent blocks
    bilstm_kernel<<<grid, 256, 0, stream>>>(x, Wih_f, Whh_f, bih_f, bhh_f,
                                            Wih_b, Whh_b, bih_b, bhh_b,
                                            (float*)d_out);
}
